// NBST_54709293417059
// MI455X (gfx1250) — matrix operand order repaired and checked by reading, not compiled, not run on hardware
//
#include <hip/hip_runtime.h>
#include <hip/hip_bf16.h>

// ---------------------------------------------------------------------------
// NBST forward for gfx1250 (MI455X). Heavy path (station ConvLSTM, ~0.35 TFLOP)
// runs on v_wmma_f32_16x16x32_f16 with LDS-resident recurrent weights.
// ---------------------------------------------------------------------------

typedef _Float16 h16;
typedef __attribute__((ext_vector_type(16))) _Float16 v16h;
typedef __attribute__((ext_vector_type(8)))  _Float16 v8h;
typedef __attribute__((ext_vector_type(8)))  float    v8f;

#define DEV __device__ __forceinline__

DEV v8f v8f_zero() {
  v8f z;
#pragma unroll
  for (int i = 0; i < 8; ++i) z[i] = 0.f;
  return z;
}
DEV v16h v16h_zero() {
  v16h z;
#pragma unroll
  for (int i = 0; i < 16; ++i) z[i] = (h16)0;
  return z;
}
DEV float sigf(float x) { return 1.f / (1.f + expf(-x)); }

// A fragment (16x32 f16): lane L holds row (L&15); K 0-15 (lanes 0-15)
// or K 16-31 (lanes 16-31).  rowp = &A[row][k0], row-major, f16.
DEV v16h a_frag(const h16* rowp, int lane) {
  int kb = (lane & 16) ? 16 : 0;
  v8h lo = *(const v8h*)(rowp + kb);
  v8h hi = *(const v8h*)(rowp + kb + 8);
  v16h r;
#pragma unroll
  for (int i = 0; i < 8; ++i) { r[i] = lo[i]; r[i + 8] = hi[i]; }
  return r;
}
// B fragment (32x16 f16) from (N,K)-major weights: lane L holds column
// n0+(L&15), K 0-15 (lanes 0-15) or 16-31 (lanes 16-31): one contiguous 32B.
DEV v16h b_frag(const h16* W, int ldk, int n0, int k0, int lane) {
  int nn = n0 + (lane & 15);
  int kb = k0 + ((lane & 16) ? 16 : 0);
  return *(const v16h*)(W + (size_t)nn * ldk + kb);
}
DEV v8f wmma_f16(v16h a, v16h b, v8f c) {
  return __builtin_amdgcn_wmma_f32_16x16x32_f16(false, a, false, b, (short)0, c,
                                                false, false);
}

// ---------------------------------------------------------------------------
// Embedding gather: station features (B,T,S,16) + 5x emb(10) -> f16 (seq,T,96)
// seq = b*64+s.  Concat order: [feat, weather, wind, hour, day, month], pad 96.
// ---------------------------------------------------------------------------
__global__ void k_embed_station(const float* __restrict__ feat,
                                const int* __restrict__ emb,
                                const float* __restrict__ wE, const float* __restrict__ wW,
                                const float* __restrict__ wM, const float* __restrict__ wD,
                                const float* __restrict__ wH,
                                h16* __restrict__ sf) {
  int idx = blockIdx.x * blockDim.x + threadIdx.x;   // (b*72+t)*64+s
  if (idx >= 128 * 72 * 64) return;
  int s = idx & 63, bt = idx >> 6;
  int t = bt % 72, b = bt / 72;
  const int* e = emb + (size_t)idx * 5;
  const float* f = feat + (size_t)idx * 16;
  h16* o = sf + ((size_t)(b * 64 + s) * 72 + t) * 96;
  for (int i = 0; i < 16; ++i) o[i] = (h16)f[i];
  const float* p;
  p = wE + e[0] * 10;       for (int j = 0; j < 10; ++j) o[16 + j] = (h16)p[j];
  p = wW + e[1] * 10;       for (int j = 0; j < 10; ++j) o[26 + j] = (h16)p[j];
  p = wH + e[4] * 10;       for (int j = 0; j < 10; ++j) o[36 + j] = (h16)p[j];
  p = wD + e[3] * 10;       for (int j = 0; j < 10; ++j) o[46 + j] = (h16)p[j];
  p = wM + (e[2] - 1) * 10; for (int j = 0; j < 10; ++j) o[56 + j] = (h16)p[j];
  for (int j = 66; j < 96; ++j) o[j] = (h16)0.f;
}

__global__ void k_embed_local(const float* __restrict__ feat,
                              const int* __restrict__ lemb,
                              const int* __restrict__ semb,
                              const float* __restrict__ wE, const float* __restrict__ wW,
                              const float* __restrict__ wM, const float* __restrict__ wD,
                              const float* __restrict__ wH,
                              h16* __restrict__ sf) {
  int idx = blockIdx.x * blockDim.x + threadIdx.x;   // b*72+t
  if (idx >= 128 * 72) return;
  h16* o = sf + (size_t)idx * 96;
  const float* f = feat + (size_t)idx * 15;
  for (int i = 0; i < 15; ++i) o[i] = (h16)f[i];
  const int* le = lemb + (size_t)idx * 2;
  const int* se = semb + (size_t)idx * 64 * 5;       // station s=0 month/day/hour
  const float* p;
  p = wE + le[0] * 10;       for (int j = 0; j < 10; ++j) o[15 + j] = (h16)p[j];
  p = wW + le[1] * 10;       for (int j = 0; j < 10; ++j) o[25 + j] = (h16)p[j];
  p = wH + se[4] * 10;       for (int j = 0; j < 10; ++j) o[35 + j] = (h16)p[j];
  p = wD + se[3] * 10;       for (int j = 0; j < 10; ++j) o[45 + j] = (h16)p[j];
  p = wM + (se[2] - 1) * 10; for (int j = 0; j < 10; ++j) o[55 + j] = (h16)p[j];
  for (int j = 65; j < 96; ++j) o[j] = (h16)0.f;
}

// ---------------------------------------------------------------------------
// Fold conv1d (3,Cin,128) into wih1 (512,128):  Wf[n][d*96+k] = conv_w[d][k][:]
// dot wih1[n][:].  Bias: xsb[n] = conv_b . wih1[n] + bih1[n] + bhh1[n].
// ---------------------------------------------------------------------------
__global__ void k_fuse_convw(const float* __restrict__ convw,
                             const float* __restrict__ wih,
                             h16* __restrict__ Wf, int Cin) {
  int idx = blockIdx.x * blockDim.x + threadIdx.x;
  if (idx >= 512 * 288) return;
  int n = idx / 288, kk = idx - n * 288;
  int d = kk / 96, k = kk - d * 96;
  float acc = 0.f;
  if (k < Cin) {
    const float* cw = convw + (size_t)(d * Cin + k) * 128;
    const float* wr = wih + (size_t)n * 128;
    for (int c = 0; c < 128; ++c) acc = fmaf(cw[c], wr[c], acc);
  }
  Wf[idx] = (h16)acc;
}

__global__ void k_fuse_bias(const float* __restrict__ convb,
                            const float* __restrict__ wih,
                            const float* __restrict__ bih,
                            const float* __restrict__ bhh,
                            float* __restrict__ xsb) {
  int n = blockIdx.x * blockDim.x + threadIdx.x;
  if (n >= 512) return;
  float a = bih[n] + bhh[n];
  const float* wr = wih + (size_t)n * 128;
  for (int c = 0; c < 128; ++c) a = fmaf(convb[c], wr[c], a);
  xsb[n] = a;
}

__global__ void k_f32_to_f16(const float* __restrict__ s, h16* __restrict__ d, int n) {
  int i = blockIdx.x * blockDim.x + threadIdx.x;
  if (i < n) d[i] = (h16)s[i];
}

// W2cat (512,256) f16 = [wih2 | whh2] along K; b2 = bih2 + bhh2
__global__ void k_w2cat(const float* __restrict__ wih2, const float* __restrict__ whh2,
                        const float* __restrict__ bih2, const float* __restrict__ bhh2,
                        h16* __restrict__ W2, float* __restrict__ b2) {
  int idx = blockIdx.x * blockDim.x + threadIdx.x;
  if (idx < 512) b2[idx] = bih2[idx] + bhh2[idx];
  if (idx >= 512 * 256) return;
  int n = idx >> 8, k = idx & 255;
  W2[idx] = (h16)((k < 128) ? wih2[(size_t)n * 128 + k] : whh2[(size_t)n * 128 + (k - 128)]);
}

// ---------------------------------------------------------------------------
// Fused conv + x-gate GEMM via WMMA:  xs[m][n] = sum_{d,k} sf[seq][t+d-1][k] *
// Wf[n][d*96+k],  m = seq*72 + t,  N = 512, K = 288 (9 chunks of 32).
// Block: 8 waves = 64(M) x 128(N) tile; wave = 16(M) x 64(N) = 4 accum tiles.
// SAME-padding handled with per-lane tap masking (t' outside [0,72) -> 0 frag).
// ---------------------------------------------------------------------------
__global__ __launch_bounds__(256) void k_xs_gemm(const h16* __restrict__ sf,
                                                 const h16* __restrict__ Wf,
                                                 h16* __restrict__ xs) {
  int wave = threadIdx.x >> 5, lane = threadIdx.x & 31;
  int m0 = blockIdx.x * 64 + (wave & 3) * 16;
  int n0 = blockIdx.y * 128 + (wave >> 2) * 64;
  int row = m0 + (lane & 15);
  int seq = row / 72;
  int t = row - seq * 72;
  v8f acc0 = v8f_zero(), acc1 = v8f_zero(), acc2 = v8f_zero(), acc3 = v8f_zero();
#pragma unroll
  for (int kc = 0; kc < 9; ++kc) {
    int d = kc / 3;
    int kk = (kc - d * 3) * 32;
    int tt = t + d - 1;
    v16h a = (tt >= 0 && tt < 72)
                 ? a_frag(sf + ((size_t)seq * 72 + tt) * 96 + kk, lane)
                 : v16h_zero();
    acc0 = wmma_f16(a, b_frag(Wf, 288, n0 +  0, kc * 32, lane), acc0);
    acc1 = wmma_f16(a, b_frag(Wf, 288, n0 + 16, kc * 32, lane), acc1);
    acc2 = wmma_f16(a, b_frag(Wf, 288, n0 + 32, kc * 32, lane), acc2);
    acc3 = wmma_f16(a, b_frag(Wf, 288, n0 + 48, kc * 32, lane), acc3);
  }
  int nc = lane & 15;
  int mofs = (lane & 16) ? 8 : 0;
#pragma unroll
  for (int r = 0; r < 8; ++r) {
    size_t mrow = (size_t)(m0 + r + mofs) * 512;
    xs[mrow + n0 +  0 + nc] = (h16)acc0[r];
    xs[mrow + n0 + 16 + nc] = (h16)acc1[r];
    xs[mrow + n0 + 32 + nc] = (h16)acc2[r];
    xs[mrow + n0 + 48 + nc] = (h16)acc3[r];
  }
}

// ---------------------------------------------------------------------------
// LSTM layer 1 (recurrent).  Workgroup owns 16 sequences x full hidden (128).
// whh f16 (512,128) staged in LDS (128 KB); h f16 in LDS.  Per step each of
// the 8 waves computes its 16-wide hidden slice of the 4 gates with 16 WMMAs,
// cell state lives in VGPRs, new h goes back to LDS + global h1 sequence.
// ---------------------------------------------------------------------------
__global__ __launch_bounds__(256) void k_lstm1(const h16* __restrict__ xs,
                                               const float* __restrict__ xsb,
                                               const h16* __restrict__ whh,
                                               h16* __restrict__ hseq) {
  __shared__ h16 w_lds[512 * 128];
  __shared__ h16 h_lds[16 * 128];
  int tid = threadIdx.x, wave = tid >> 5, lane = tid & 31;
  for (int i = tid; i < 512 * 128 / 8; i += 256)
    ((v8h*)w_lds)[i] = ((const v8h*)whh)[i];
  for (int i = tid; i < 16 * 128 / 8; i += 256) {
    v8h z;
#pragma unroll
    for (int j = 0; j < 8; ++j) z[j] = (h16)0;
    ((v8h*)h_lds)[i] = z;
  }
  __syncthreads();
  int seq0 = blockIdx.x * 16;
  int hs = wave;
  int nc = lane & 15;
  int hid = hs * 16 + nc;
  int mofs = (lane & 16) ? 8 : 0;
  float xb0 = xsb[hid], xb1 = xsb[128 + hid], xb2 = xsb[256 + hid], xb3 = xsb[384 + hid];
  v8f c = v8f_zero();
  for (int t = 0; t < 72; ++t) {
    if (t < 71 && lane == 0)
      __builtin_prefetch(xs + ((size_t)seq0 * 72 + t + 1) * 512, 0, 1);
    v8f g0 = v8f_zero(), g1 = v8f_zero(), g2 = v8f_zero(), g3 = v8f_zero();
#pragma unroll
    for (int kc = 0; kc < 4; ++kc) {
      v16h a = a_frag(h_lds + (lane & 15) * 128 + kc * 32, lane);
      g0 = wmma_f16(a, b_frag(w_lds, 128,   0 + hs * 16, kc * 32, lane), g0);
      g1 = wmma_f16(a, b_frag(w_lds, 128, 128 + hs * 16, kc * 32, lane), g1);
      g2 = wmma_f16(a, b_frag(w_lds, 128, 256 + hs * 16, kc * 32, lane), g2);
      g3 = wmma_f16(a, b_frag(w_lds, 128, 384 + hs * 16, kc * 32, lane), g3);
    }
    __syncthreads();                       // everyone done reading h(t-1)
#pragma unroll
    for (int r = 0; r < 8; ++r) {
      int m = r + mofs;
      size_t xrow = ((size_t)(seq0 + m) * 72 + t) * 512;
      float gi = g0[r] + (float)xs[xrow + hid] + xb0;
      float gf = g1[r] + (float)xs[xrow + 128 + hid] + xb1;
      float gc = g2[r] + (float)xs[xrow + 256 + hid] + xb2;
      float go = g3[r] + (float)xs[xrow + 384 + hid] + xb3;
      float cc = sigf(gf) * c[r] + sigf(gi) * tanhf(gc);
      c[r] = cc;
      float hh = sigf(go) * tanhf(cc);
      h_lds[m * 128 + hid] = (h16)hh;
      hseq[((size_t)(seq0 + m) * 72 + t) * 128 + hid] = (h16)hh;
    }
    __syncthreads();                       // h(t) visible for next step
  }
}

// ---------------------------------------------------------------------------
// LSTM layer 2: x-gates computed on the fly: g = [h1(t) ; h(t-1)] @ [wih2;whh2]^T
// whh2 half of W2cat staged in LDS, wih2 half read from global.  Only the
// final h (t=71) is stored (f32).
// ---------------------------------------------------------------------------
__global__ __launch_bounds__(256) void k_lstm2(const h16* __restrict__ h1,
                                               const h16* __restrict__ W2,
                                               const float* __restrict__ b2,
                                               float* __restrict__ hout) {
  __shared__ h16 w_lds[512 * 128];
  __shared__ h16 h_lds[16 * 128];
  int tid = threadIdx.x, wave = tid >> 5, lane = tid & 31;
  for (int i = tid; i < 512 * 16; i += 256) {
    int rowi = i >> 4, c8 = i & 15;
    *(v8h*)(w_lds + rowi * 128 + c8 * 8) = *(const v8h*)(W2 + (size_t)rowi * 256 + 128 + c8 * 8);
  }
  for (int i = tid; i < 16 * 128 / 8; i += 256) {
    v8h z;
#pragma unroll
    for (int j = 0; j < 8; ++j) z[j] = (h16)0;
    ((v8h*)h_lds)[i] = z;
  }
  __syncthreads();
  int seq0 = blockIdx.x * 16;
  int hs = wave;
  int nc = lane & 15;
  int hid = hs * 16 + nc;
  int mofs = (lane & 16) ? 8 : 0;
  float xb0 = b2[hid], xb1 = b2[128 + hid], xb2 = b2[256 + hid], xb3 = b2[384 + hid];
  v8f c = v8f_zero();
  for (int t = 0; t < 72; ++t) {
    v8f g0 = v8f_zero(), g1 = v8f_zero(), g2 = v8f_zero(), g3 = v8f_zero();
#pragma unroll
    for (int kc = 0; kc < 4; ++kc) {   // x part: h1(t) vs wih2 (global)
      v16h a = a_frag(h1 + ((size_t)(seq0 + (lane & 15)) * 72 + t) * 128 + kc * 32, lane);
      g0 = wmma_f16(a, b_frag(W2, 256,   0 + hs * 16, kc * 32, lane), g0);
      g1 = wmma_f16(a, b_frag(W2, 256, 128 + hs * 16, kc * 32, lane), g1);
      g2 = wmma_f16(a, b_frag(W2, 256, 256 + hs * 16, kc * 32, lane), g2);
      g3 = wmma_f16(a, b_frag(W2, 256, 384 + hs * 16, kc * 32, lane), g3);
    }
#pragma unroll
    for (int kc = 0; kc < 4; ++kc) {   // recurrent part: h(t-1) vs whh2 (LDS)
      v16h a = a_frag(h_lds + (lane & 15) * 128 + kc * 32, lane);
      g0 = wmma_f16(a, b_frag(w_lds, 128,   0 + hs * 16, kc * 32, lane), g0);
      g1 = wmma_f16(a, b_frag(w_lds, 128, 128 + hs * 16, kc * 32, lane), g1);
      g2 = wmma_f16(a, b_frag(w_lds, 128, 256 + hs * 16, kc * 32, lane), g2);
      g3 = wmma_f16(a, b_frag(w_lds, 128, 384 + hs * 16, kc * 32, lane), g3);
    }
    __syncthreads();
#pragma unroll
    for (int r = 0; r < 8; ++r) {
      int m = r + mofs;
      float cc = sigf(g1[r] + xb1) * c[r] + sigf(g0[r] + xb0) * tanhf(g2[r] + xb2);
      c[r] = cc;
      float hh = sigf(g3[r] + xb3) * tanhf(cc);
      h_lds[m * 128 + hid] = (h16)hh;
      if (t == 71) hout[(size_t)(seq0 + m) * 128 + hid] = hh;
    }
    __syncthreads();
  }
}

// ---------------------------------------------------------------------------
// Small f32 kernels for the attention / MLP / pred tail (negligible FLOPs).
// ---------------------------------------------------------------------------
__global__ void k_sgemm(const float* __restrict__ A, int lda,
                        const float* __restrict__ W, const float* __restrict__ bias,
                        float* __restrict__ C, int ldc,
                        int rows, int N, int K, int act) {
  int idx = blockIdx.x * blockDim.x + threadIdx.x;
  if (idx >= rows * N) return;
  int r = idx / N, n = idx - r * N;
  const float* a = A + (size_t)r * lda;
  const float* w = W + (size_t)n * K;
  float acc = bias ? bias[n] : 0.f;
  for (int k = 0; k < K; ++k) acc = fmaf(a[k], w[k], acc);
  if (act == 1) acc = fmaxf(acc, 0.f);
  else if (act == 2) acc = acc / (1.f + expf(-acc));                       // silu
  else if (act == 3) acc = 0.5f * acc * (1.f + erff(acc * 0.7071067811865475f)); // gelu
  C[(size_t)r * ldc + n] = acc;
}

__global__ void k_softmax(float* __restrict__ X, int rows, int len) {
  int r = blockIdx.x * blockDim.x + threadIdx.x;
  if (r >= rows) return;
  float* x = X + (size_t)r * len;
  float mx = x[0];
  for (int i = 1; i < len; ++i) mx = fmaxf(mx, x[i]);
  float s = 0.f;
  for (int i = 0; i < len; ++i) { float e = expf(x[i] - mx); x[i] = e; s += e; }
  float inv = 1.f / s;
  for (int i = 0; i < len; ++i) x[i] *= inv;
}

__global__ void k_layernorm(const float* __restrict__ X, const float* __restrict__ g,
                            const float* __restrict__ b, float* __restrict__ Y, int rows) {
  int r = blockIdx.x * blockDim.x + threadIdx.x;
  if (r >= rows) return;
  const float* x = X + (size_t)r * 128;
  float m = 0.f;
  for (int i = 0; i < 128; ++i) m += x[i];
  m *= (1.f / 128.f);
  float v = 0.f;
  for (int i = 0; i < 128; ++i) { float d = x[i] - m; v = fmaf(d, d, v); }
  v *= (1.f / 128.f);
  float inv = rsqrtf(v + 1e-5f);
  for (int i = 0; i < 128; ++i) Y[(size_t)r * 128 + i] = (x[i] - m) * inv * g[i] + b[i];
}

// one thread per (b, head): 1 query x 65 keys, softmax, weighted sum + alpha*q
__global__ void k_attn(const float* __restrict__ qh, const float* __restrict__ kh,
                       const float* __restrict__ vh, const float* __restrict__ alpha,
                       float* __restrict__ outp) {
  int idx = blockIdx.x * blockDim.x + threadIdx.x;
  if (idx >= 128 * 8) return;
  int b = idx >> 3, hd = idx & 7;
  const float* q = qh + (size_t)b * 128 + hd * 16;
  float sc[65];
  float mx = -1e30f;
  for (int s = 0; s < 65; ++s) {
    const float* kp = kh + ((size_t)b * 65 + s) * 128 + hd * 16;
    float d = 0.f;
    for (int j = 0; j < 16; ++j) d = fmaf(q[j], kp[j], d);
    d *= 0.08838834764831845f;   // 128^-0.5
    sc[s] = d;
    mx = fmaxf(mx, d);
  }
  float sum = 0.f;
  for (int s = 0; s < 65; ++s) { sc[s] = expf(sc[s] - mx); sum += sc[s]; }
  float inv = 1.f / sum;
  float al = alpha[hd];
  for (int j = 0; j < 16; ++j) {
    float o = 0.f;
    for (int s = 0; s < 65; ++s)
      o = fmaf(sc[s], vh[((size_t)b * 65 + s) * 128 + hd * 16 + j], o);
    outp[(size_t)b * 128 + hd * 16 + j] = o * inv + al * q[j];
  }
}

__global__ void k_gather_nodes(const float* __restrict__ local_node,
                               const float* __restrict__ station_nodes,
                               float* __restrict__ X) {
  int idx = blockIdx.x * blockDim.x + threadIdx.x;
  if (idx >= 8320 * 16) return;
  int r = idx >> 4, c = idx & 15;
  int b = r / 65, j = r - b * 65;
  X[idx] = (j == 0) ? local_node[(size_t)b * 16 + c]
                    : station_nodes[((size_t)b * 64 + (j - 1)) * 16 + c];
}

__global__ void k_build_dyn_nodes(const float* __restrict__ loc_h,
                                  const float* __restrict__ st_h,
                                  float* __restrict__ nodes) {
  int idx = blockIdx.x * blockDim.x + threadIdx.x;
  if (idx >= 8320 * 128) return;
  int r = idx >> 7, c = idx & 127;
  int b = r / 65, j = r - b * 65;
  nodes[idx] = (j == 0) ? loc_h[(size_t)b * 128 + c]
                        : st_h[((size_t)b * 64 + (j - 1)) * 128 + c];
}

__global__ void k_cur_init(const float* __restrict__ src, int srcStride,
                           float* __restrict__ cur, float* __restrict__ outs) {
  int idx = blockIdx.x * blockDim.x + threadIdx.x;
  if (idx >= 128 * 128) return;
  int b = idx >> 7, c = idx & 127;
  float v = src[(size_t)b * srcStride + c];
  cur[idx] = v;
  outs[(size_t)b * 512 + c] = v;
}

__global__ void k_copy_outs(const float* __restrict__ cur, float* __restrict__ outs, int col) {
  int idx = blockIdx.x * blockDim.x + threadIdx.x;
  if (idx >= 128 * 128) return;
  int b = idx >> 7, c = idx & 127;
  outs[(size_t)b * 512 + col + c] = cur[idx];
}

// Sequential pred scan: one block (128 threads) per batch row, 72 steps.
__global__ __launch_bounds__(128) void k_pred(const float* __restrict__ base,
                                              const float* __restrict__ w1,
                                              const float* __restrict__ b1,
                                              const float* __restrict__ w2,
                                              const float* __restrict__ b2,
                                              const float* __restrict__ sigma_h,
                                              float* __restrict__ out) {
  __shared__ float xb[256];
  __shared__ float act[128];
  __shared__ float hv;
  int b = blockIdx.x, tid = threadIdx.x;
  xb[tid] = base[(size_t)b * 256 + tid];
  xb[tid + 128] = base[(size_t)b * 256 + 128 + tid];
  if (tid == 0) hv = sigma_h[0];
  __syncthreads();
  const float* wr = w1 + (size_t)tid * 257;
  for (int t = 0; t < 72; ++t) {
    float a = b1[tid] + wr[256] * hv;
    for (int k = 0; k < 256; ++k) a = fmaf(wr[k], xb[k], a);
    act[tid] = fmaxf(a, 0.f);
    __syncthreads();
    if (tid == 0) {
      float o = b2[0];
      for (int k = 0; k < 128; ++k) o = fmaf(w2[k], act[k], o);
      o = tanhf(o);
      out[(size_t)b * 72 + t] = o;
      hv = o;
    }
    __syncthreads();
  }
}

// ---------------------------------------------------------------------------
// Host orchestration.  Param leaves follow jax tree order (dicts sorted).
// ---------------------------------------------------------------------------
extern "C" void kernel_launch(void* const* d_in, const int* in_sizes, int n_in,
                              void* d_out, int out_size, void* d_ws, size_t ws_size,
                              hipStream_t stream) {
  (void)in_sizes; (void)n_in; (void)out_size; (void)ws_size;
#define PF(i) ((const float*)d_in[i])
#define PI(i) ((const int*)d_in[i])
  // inputs
  const float* local_node       = PF(1);
  const float* local_features   = PF(2);
  const int*   local_emb        = PI(3);
  const float* station_nodes    = PF(4);
  const float* station_features = PF(5);
  const int*   station_emb      = PI(6);
  // param leaf indices (base 7): day_emb=7; dyn_att{0,1}.{b,w}=8..11;
  // dyn_fus{0,1}.{b,w}=12..15; dyn_layer[l] base 16+13l
  // (alpha,ff.b,ff.w,k.b,k.w,ln.b,ln.g,out1.b,out1.w,q.b,q.w,v.b,v.w);
  // hour=55; loc_lstm: conv_b=56 conv_w=57 l0{bhh,bih,whh,wih}=58..61
  // l1=62..65; month=66; pred1{b,w}=67,68; pred2{b,w}=69,70; sigma_h=71;
  // st_att=72..75; st_fus=76..79; st_layer base 80+13l; smlp=119..122;
  // st_lstm: conv_b=123 conv_w=124 l0=125..128 l1=129..132; weather=133; wind=134.

  // ---- workspace carve-up ----
  size_t off = 0;
  auto alloc = [&](size_t bytes) -> void* {
    void* p = (char*)d_ws + off;
    off = (off + bytes + 255) & ~(size_t)255;
    return p;
  };
  h16* sf_st   = (h16*)alloc((size_t)8192 * 72 * 96 * 2);
  h16* sf_loc  = (h16*)alloc((size_t)128 * 72 * 96 * 2);
  h16* wf_st   = (h16*)alloc(512 * 288 * 2);
  h16* wf_loc  = (h16*)alloc(512 * 288 * 2);
  float* xsb_st  = (float*)alloc(512 * 4);
  float* xsb_loc = (float*)alloc(512 * 4);
  h16* whh1_st  = (h16*)alloc(512 * 128 * 2);
  h16* whh1_loc = (h16*)alloc(512 * 128 * 2);
  h16* w2_st    = (h16*)alloc(512 * 256 * 2);
  h16* w2_loc   = (h16*)alloc(512 * 256 * 2);
  float* b2_st  = (float*)alloc(512 * 4);
  float* b2_loc = (float*)alloc(512 * 4);
  h16* xs_st  = (h16*)alloc((size_t)589824 * 512 * 2);
  h16* xs_loc = (h16*)alloc((size_t)9216 * 512 * 2);
  h16* h1_st  = (h16*)alloc((size_t)589824 * 128 * 2);
  h16* h1_loc = (h16*)alloc((size_t)9216 * 128 * 2);
  float* st_h  = (float*)alloc((size_t)8192 * 128 * 4);
  float* loc_h = (float*)alloc((size_t)128 * 128 * 4);
  float* Xn   = (float*)alloc((size_t)8320 * 16 * 4);
  float* tbuf = (float*)alloc((size_t)8320 * 256 * 4);
  float* embn = (float*)alloc((size_t)8320 * 128 * 4);
  float* fusb = (float*)alloc((size_t)8320 * 128 * 4);
  float* kbuf = (float*)alloc((size_t)8320 * 128 * 4);
  float* vbuf = (float*)alloc((size_t)8320 * 128 * 4);
  float* qbuf = (float*)alloc((size_t)128 * 128 * 4);
  float* attb = (float*)alloc((size_t)128 * 128 * 4);
  float* o1b  = (float*)alloc((size_t)128 * 128 * 4);
  float* ffb  = (float*)alloc((size_t)128 * 128 * 4);
  float* curb = (float*)alloc((size_t)128 * 128 * 4);
  float* outs = (float*)alloc((size_t)128 * 512 * 4);
  float* fh   = (float*)alloc((size_t)128 * 256 * 4);
  float* baseb = (float*)alloc((size_t)128 * 256 * 4);

  auto sgemm = [&](const float* A, int lda, const float* W, const float* bias,
                   float* C, int ldc, int rows, int N, int K, int act) {
    int total = rows * N;
    k_sgemm<<<(total + 255) / 256, 256, 0, stream>>>(A, lda, W, bias, C, ldc,
                                                     rows, N, K, act);
  };

  // ---- ConvLSTM heavy path (WMMA) ----
  k_embed_station<<<(589824 + 255) / 256, 256, 0, stream>>>(
      station_features, station_emb, PF(133), PF(134), PF(66), PF(7), PF(55), sf_st);
  k_embed_local<<<(9216 + 255) / 256, 256, 0, stream>>>(
      local_features, local_emb, station_emb, PF(133), PF(134), PF(66), PF(7), PF(55), sf_loc);

  k_fuse_convw<<<(512 * 288 + 255) / 256, 256, 0, stream>>>(PF(124), PF(128), wf_st, 66);
  k_fuse_convw<<<(512 * 288 + 255) / 256, 256, 0, stream>>>(PF(57), PF(61), wf_loc, 65);
  k_fuse_bias<<<2, 256, 0, stream>>>(PF(123), PF(128), PF(126), PF(125), xsb_st);
  k_fuse_bias<<<2, 256, 0, stream>>>(PF(56), PF(61), PF(59), PF(58), xsb_loc);
  k_f32_to_f16<<<256, 256, 0, stream>>>(PF(127), whh1_st, 512 * 128);
  k_f32_to_f16<<<256, 256, 0, stream>>>(PF(60), whh1_loc, 512 * 128);
  k_w2cat<<<512, 256, 0, stream>>>(PF(132), PF(131), PF(130), PF(129), w2_st, b2_st);
  k_w2cat<<<512, 256, 0, stream>>>(PF(65), PF(64), PF(63), PF(62), w2_loc, b2_loc);

  { dim3 g(9216, 4); k_xs_gemm<<<g, 256, 0, stream>>>(sf_st, wf_st, xs_st); }
  { dim3 g(144, 4);  k_xs_gemm<<<g, 256, 0, stream>>>(sf_loc, wf_loc, xs_loc); }

  k_lstm1<<<512, 256, 0, stream>>>(xs_st, xsb_st, whh1_st, h1_st);
  k_lstm1<<<8,   256, 0, stream>>>(xs_loc, xsb_loc, whh1_loc, h1_loc);
  k_lstm2<<<512, 256, 0, stream>>>(h1_st, w2_st, b2_st, st_h);
  k_lstm2<<<8,   256, 0, stream>>>(h1_loc, w2_loc, b2_loc, loc_h);

  // ---- attention stack (shared by static / dynamic paths) ----
  auto attn_stack = [&](int pbase, const float* kv, float* cur, float* outsb) {
    for (int l = 0; l < 3; ++l) {
      int pb = pbase + 13 * l;
      sgemm(cur, 128, PF(pb + 10), PF(pb + 9), qbuf, 128, 128, 128, 128, 0);   // q
      sgemm(kv, 128, PF(pb + 4), PF(pb + 3), kbuf, 128, 8320, 128, 128, 0);    // k
      sgemm(kv, 128, PF(pb + 12), PF(pb + 11), vbuf, 128, 8320, 128, 128, 0);  // v
      k_attn<<<4, 256, 0, stream>>>(qbuf, kbuf, vbuf, PF(pb + 0), attb);
      sgemm(attb, 128, PF(pb + 8), PF(pb + 7), o1b, 128, 128, 128, 128, 0);    // out1
      k_layernorm<<<1, 128, 0, stream>>>(o1b, PF(pb + 6), PF(pb + 5), ffb, 128);
      sgemm(ffb, 128, PF(pb + 2), PF(pb + 1), cur, 128, 128, 128, 128, 3);     // ff+gelu
      k_copy_outs<<<64, 256, 0, stream>>>(cur, outsb, 128 * (l + 1));
    }
  };

  // ---- static path ----
  k_gather_nodes<<<(8320 * 16 + 255) / 256, 256, 0, stream>>>(local_node, station_nodes, Xn);
  sgemm(Xn, 16, PF(120), PF(119), tbuf, 256, 8320, 256, 16, 2);     // smlp0 + silu
  sgemm(tbuf, 256, PF(122), PF(121), embn, 128, 8320, 128, 256, 2); // smlp1 + silu
  k_cur_init<<<64, 256, 0, stream>>>(embn, 65 * 128, curb, outs);
  sgemm(embn, 128, PF(73), PF(72), tbuf, 128, 8320, 128, 128, 1);   // st_att0 + relu
  sgemm(tbuf, 128, PF(75), PF(74), fusb, 128, 8320, 128, 128, 0);   // st_att1
  k_softmax<<<(8320 + 255) / 256, 256, 0, stream>>>(fusb, 8320, 128);
  attn_stack(80, fusb, curb, outs);
  sgemm(outs, 512, PF(77), PF(76), fh, 256, 128, 256, 512, 1);      // st_fus0 + relu
  sgemm(fh, 256, PF(79), PF(78), baseb, 256, 128, 128, 256, 1);     // -> base[:, :128]

  // ---- dynamic path ----
  k_build_dyn_nodes<<<(8320 * 128 + 255) / 256, 256, 0, stream>>>(loc_h, st_h, embn);
  k_cur_init<<<64, 256, 0, stream>>>(loc_h, 128, curb, outs);
  sgemm(embn, 128, PF(9), PF(8), tbuf, 128, 8320, 128, 128, 1);     // dyn_att0 + relu
  sgemm(tbuf, 128, PF(11), PF(10), fusb, 128, 8320, 128, 128, 0);   // dyn_att1
  k_softmax<<<(8320 + 255) / 256, 256, 0, stream>>>(fusb, 8320, 128);
  attn_stack(16, fusb, curb, outs);
  sgemm(outs, 512, PF(13), PF(12), fh, 256, 128, 256, 512, 1);      // dyn_fus0 + relu
  sgemm(fh, 256, PF(15), PF(14), baseb + 128, 256, 128, 128, 256, 1); // -> base[:, 128:]

  // ---- autoregressive prediction head ----
  k_pred<<<128, 128, 0, stream>>>(baseb, PF(68), PF(67), PF(70), PF(69), PF(71),
                                  (float*)d_out);
#undef PF
#undef PI
}
